// LinearCustom_28063316312489
// MI455X (gfx1250) — matrix operand order repaired and checked by reading, not compiled, not run on hardware
//
#include <hip/hip_runtime.h>
#include <stdint.h>

// ---------------------------------------------------------------------------
// Quantized linear (dequant int4-in-int8 groups -> f16) + GEMM + bias
//   M=16384 (8*2048), N=4096 (out), K=4096 (in)
// Path: f16 WMMA (v_wmma_f32_16x16x32_f16), async global->LDS double buffer.
// Workspace: XH (M*K f16 = 128MB) + WH (N*K f16 = 32MB) = 160MB of d_ws.
// ---------------------------------------------------------------------------

typedef __attribute__((ext_vector_type(16))) _Float16 v16h;
typedef __attribute__((ext_vector_type(8)))  float    v8f;

#define M_TOT 16384
#define N_TOT 4096
#define K_TOT 4096

#define BM 128
#define BN 128
#define BK 32
#define LDT 40                 // padded LDS row stride (halves): 64-bank friendly
#define TILE_H (128 * LDT)

// ---- CDNA5 async copy: global -> LDS, tracked with ASYNCcnt ----
static __device__ __forceinline__ void async_ld16(const _Float16* gp, _Float16* lp) {
  asm volatile("global_load_async_to_lds_b128 %0, %1, off"
               :
               : "v"((uint32_t)(uintptr_t)lp), "v"((uint64_t)(uintptr_t)gp)
               : "memory");
}

#if __has_builtin(__builtin_amdgcn_s_wait_asynccnt)
#define WAIT_ASYNC(n) __builtin_amdgcn_s_wait_asynccnt(n)
#else
#define WAIT_ASYNC(n) asm volatile("s_wait_asynccnt " #n ::: "memory")
#endif

union FragH { uint4 u[2]; v16h v; };

// ---------------------------------------------------------------------------
// Prepass 1: x f32 -> f16
// ---------------------------------------------------------------------------
__global__ __launch_bounds__(256) void cvt_x_f16(const float* __restrict__ x,
                                                 _Float16* __restrict__ xh) {
  size_t i = ((size_t)blockIdx.x * 256 + threadIdx.x) * 4;
  float4 v = *(const float4*)(x + i);
  union { _Float16 h[4]; uint2 u; } o;
  o.h[0] = (_Float16)v.x; o.h[1] = (_Float16)v.y;
  o.h[2] = (_Float16)v.z; o.h[3] = (_Float16)v.w;
  *(uint2*)(xh + i) = o.u;
}

// ---------------------------------------------------------------------------
// Prepass 2: dequant  w = (q/15) * group_scale  -> f16 [out][in] row-major
// ---------------------------------------------------------------------------
__global__ __launch_bounds__(256) void dequant_w(const int8_t* __restrict__ q,
                                                 const _Float16* __restrict__ norm,
                                                 _Float16* __restrict__ wh) {
  size_t g = (size_t)blockIdx.x * 256 + threadIdx.x;   // one group of 16 per thread
  union { int4 v; int8_t b[16]; } qq;
  qq.v = *(const int4*)(q + g * 16);
  const float s = (float)norm[g] * (1.0f / 15.0f);
  union { _Float16 h[16]; uint4 u[2]; } o;
#pragma unroll
  for (int i = 0; i < 16; ++i) o.h[i] = (_Float16)((float)qq.b[i] * s);
  uint4* dst = (uint4*)(wh + g * 16);
  dst[0] = o.u[0];
  dst[1] = o.u[1];
}

// ---------------------------------------------------------------------------
// GEMM: out[m][n] = XH[m][:] . WH[n][:] + bias[n]
// Block = 128x128, 8 waves; wave = 64x32 (4x2 fragments of 16x16).
// ---------------------------------------------------------------------------
__global__ __launch_bounds__(256) void gemm_f16_wmma(
    const _Float16* __restrict__ XH, const _Float16* __restrict__ WH,
    const float* __restrict__ bias, float* __restrict__ out) {
  __shared__ __align__(16) _Float16 As[2][TILE_H];   // 20KB
  __shared__ __align__(16) _Float16 Bs[2][TILE_H];   // 20KB

  const int tid  = threadIdx.x;
  const int lane = tid & 31;
  const int wave = tid >> 5;
  const int wr = wave >> 2;            // 0..1 : 64-row slab
  const int wc = wave & 3;             // 0..3 : 32-col slab
  const int m0 = blockIdx.y * BM;
  const int n0 = blockIdx.x * BN;

  // staging: each thread moves 2x16B of A and 2x16B of B per BK step
  const int srow = tid >> 2;           // 0..63 (and +64)
  const int scol = (tid & 3) * 8;      // halves within the BK=32 slice
  const _Float16* gA = XH + (size_t)(m0 + srow) * K_TOT + scol;
  const _Float16* gB = WH + (size_t)(n0 + srow) * K_TOT + scol;
  const int loff = srow * LDT + scol;

  v8f acc[4][2] = {};

  // fragment lane addressing per ISA 7.12.2 (16-bit A 16x32, B 32x16)
  const int arow = wr * 64 + (lane & 15);
  const int akh  = (lane >> 4) * 8;    // halves 0..7 then +16.. per layout
  const int brow = wc * 32 + (lane & 15);
  const int bkh  = (lane >> 4) * 8;    // halves 0..7 then +16.. per layout

  // prologue: stage 0
  async_ld16(gA,                        &As[0][loff]);
  async_ld16(gA + (size_t)64 * K_TOT,   &As[0][loff + 64 * LDT]);
  async_ld16(gB,                        &Bs[0][loff]);
  async_ld16(gB + (size_t)64 * K_TOT,   &Bs[0][loff + 64 * LDT]);

  for (int k0 = 0; k0 < K_TOT; k0 += BK) {
    const int cur = (k0 / BK) & 1;
    if (k0 + BK < K_TOT) {
      const int nxt = cur ^ 1;
      const _Float16* ga = gA + k0 + BK;
      const _Float16* gb = gB + k0 + BK;
      async_ld16(ga,                      &As[nxt][loff]);
      async_ld16(ga + (size_t)64 * K_TOT, &As[nxt][loff + 64 * LDT]);
      async_ld16(gb,                      &Bs[nxt][loff]);
      async_ld16(gb + (size_t)64 * K_TOT, &Bs[nxt][loff + 64 * LDT]);
      WAIT_ASYNC(4);                     // stage `cur` (4 older ops) complete
    } else {
      WAIT_ASYNC(0);
    }
    __syncthreads();

    v16h a[4], b[2];
#pragma unroll
    for (int fi = 0; fi < 4; ++fi) {
      const _Float16* p = &As[cur][(arow + fi * 16) * LDT + akh];
      FragH f;
      f.u[0] = *(const uint4*)(p);        // K = kb .. kb+7
      f.u[1] = *(const uint4*)(p + 16);   // K = kb+16 .. kb+23
      a[fi] = f.v;
    }
#pragma unroll
    for (int fj = 0; fj < 2; ++fj) {
      const _Float16* p = &Bs[cur][(brow + fj * 16) * LDT + bkh];
      FragH f;
      f.u[0] = *(const uint4*)(p);        // K = kb .. kb+7
      f.u[1] = *(const uint4*)(p + 16);   // K = kb+16 .. kb+23
      b[fj] = f.v;
    }
#pragma unroll
    for (int fi = 0; fi < 4; ++fi)
#pragma unroll
      for (int fj = 0; fj < 2; ++fj)
        acc[fi][fj] = __builtin_amdgcn_wmma_f32_16x16x32_f16(
            false, a[fi], false, b[fj], (short)0, acc[fi][fj], false, false);

    __syncthreads();   // everyone done reading `cur` before it is refilled
  }

  // epilogue: bias + store (C layout: n = lane%16, m = (lane>=16)*8 + vgpr)
#pragma unroll
  for (int fj = 0; fj < 2; ++fj) {
    const int n = n0 + wc * 32 + fj * 16 + (lane & 15);
    const float bv = bias[n];
#pragma unroll
    for (int fi = 0; fi < 4; ++fi) {
      const int mb = m0 + wr * 64 + fi * 16 + ((lane >> 4) * 8);
#pragma unroll
      for (int r = 0; r < 8; ++r)
        out[(size_t)(mb + r) * N_TOT + n] = acc[fi][fj][r] + bv;
    }
  }
}

// ---------------------------------------------------------------------------
extern "C" void kernel_launch(void* const* d_in, const int* in_sizes, int n_in,
                              void* d_out, int out_size, void* d_ws, size_t ws_size,
                              hipStream_t stream) {
  const float*    x  = (const float*)d_in[0];
  const int8_t*   wq = (const int8_t*)d_in[1];
  const _Float16* wn = (const _Float16*)d_in[2];
  const float*    bs = (const float*)d_in[3];
  float*          out = (float*)d_out;

  _Float16* XH = (_Float16*)d_ws;                         // 128 MB
  _Float16* WH = XH + (size_t)M_TOT * K_TOT;              // +32 MB

  // 1) x -> f16
  cvt_x_f16<<<(unsigned)((size_t)M_TOT * K_TOT / 4 / 256), 256, 0, stream>>>(x, XH);
  // 2) dequant weights -> f16 [out][in]
  dequant_w<<<(unsigned)((size_t)N_TOT * K_TOT / 16 / 256), 256, 0, stream>>>(wq, wn, WH);
  // 3) WMMA GEMM + bias (N fastest so an A-stripe is L2-reused across 32 blocks)
  dim3 grid(N_TOT / BN, M_TOT / BM);
  gemm_f16_wmma<<<grid, 256, 0, stream>>>(XH, WH, bs, out);

  (void)in_sizes; (void)n_in; (void)out_size; (void)ws_size;
}
